// H2GCN_Net_36945308680872
// MI455X (gfx1250) — matrix operand order repaired and checked by reading, not compiled, not run on hardware
//
#include <hip/hip_runtime.h>

// ---------------------------------------------------------------------------
// H2GCN-style GNN forward on MI455X (gfx1250, wave32).
//  - Dense GEMMs via v_wmma_f32_16x16x32_bf16 (fp32 accum), one wave per
//    16-row stripe computing ALL column tiles (A fragment reused).
//  - f32->bf16 via native converts (plain casts -> v_cvt_pk_bf16_f32).
//  - Edge gather/scatter via float atomics (the HBM-bound part).
// ---------------------------------------------------------------------------

#define N_NODES 100000
#define N_EDGES 1600000
#define IN_DIM  128
#define HID     64
#define OUTD    40
#define OUTP    48   // padded to 3 column tiles of 16

typedef __bf16 v16bf __attribute__((ext_vector_type(16)));
typedef float  v8f   __attribute__((ext_vector_type(8)));

union Frag16 {
    v16bf v;
    unsigned short u[16];
    uint4 q[2];
};

__device__ __forceinline__ __bf16 tobf(float f) { return (__bf16)f; }

// round-to-nearest-even fallback used only in weight prep (cold path)
__device__ __forceinline__ unsigned short f2bf_bits(float f) {
    unsigned u = __float_as_uint(f);
    unsigned r = u + 0x7FFFu + ((u >> 16) & 1u);
    return (unsigned short)(r >> 16);
}

// Build a 16-element bf16 A fragment from 16 fp32 values:
// elements 0..7  = A[row, kb..kb+7],  elements 8..15 = A[row, kb+16..kb+23]
__device__ __forceinline__ v16bf load_a_frag(const float* __restrict__ arow, int kb) {
    const float4* p0 = (const float4*)(arow + kb);
    const float4* p1 = (const float4*)(arow + kb + 16);
    float4 f0 = p0[0], f1 = p0[1], f2 = p1[0], f3 = p1[1];
    Frag16 a;
    a.v[0]  = tobf(f0.x); a.v[1]  = tobf(f0.y); a.v[2]  = tobf(f0.z); a.v[3]  = tobf(f0.w);
    a.v[4]  = tobf(f1.x); a.v[5]  = tobf(f1.y); a.v[6]  = tobf(f1.z); a.v[7]  = tobf(f1.w);
    a.v[8]  = tobf(f2.x); a.v[9]  = tobf(f2.y); a.v[10] = tobf(f2.z); a.v[11] = tobf(f2.w);
    a.v[12] = tobf(f3.x); a.v[13] = tobf(f3.y); a.v[14] = tobf(f3.z); a.v[15] = tobf(f3.w);
    return a.v;
}

// ---------------------------------------------------------------------------
// Weight prep: fp32 row-major [K,N] -> bf16 column-major [N,K] (B-fragment
// friendly: per output column, K is contiguous).  Wf padded 40->48 cols.
// ---------------------------------------------------------------------------
__global__ void prep_weights(const float* __restrict__ W1,
                             const float* __restrict__ Wc1,
                             const float* __restrict__ Wc2,
                             const float* __restrict__ Wf,
                             unsigned short* __restrict__ W1t,
                             unsigned short* __restrict__ Wc1t,
                             unsigned short* __restrict__ Wc2t,
                             unsigned short* __restrict__ Wft) {
    int t = blockIdx.x * blockDim.x + threadIdx.x;
    if (t < HID * IN_DIM) {                       // W1t[n][k], n<64, k<128
        int n = t / IN_DIM, k = t % IN_DIM;
        W1t[t] = f2bf_bits(W1[k * HID + n]);
        return;
    }
    t -= HID * IN_DIM;
    if (t < HID * HID) {
        int n = t / HID, k = t % HID;
        Wc1t[t] = f2bf_bits(Wc1[k * HID + n]);
        return;
    }
    t -= HID * HID;
    if (t < HID * HID) {
        int n = t / HID, k = t % HID;
        Wc2t[t] = f2bf_bits(Wc2[k * HID + n]);
        return;
    }
    t -= HID * HID;
    if (t < OUTP * (3 * HID)) {                   // Wft[n][k], n<48, k<192
        int n = t / (3 * HID), k = t % (3 * HID);
        Wft[t] = (n < OUTD) ? f2bf_bits(Wf[k * OUTD + n]) : (unsigned short)0;
        return;
    }
}

// ---------------------------------------------------------------------------
// Degree / normalization helpers
// ---------------------------------------------------------------------------
__global__ void deg_zero(float* __restrict__ deg) {
    int t = blockIdx.x * blockDim.x + threadIdx.x;
    if (t < N_NODES) deg[t] = 0.0f;
}

__global__ void deg_count(const int* __restrict__ col, float* __restrict__ deg) {
    int t = blockIdx.x * blockDim.x + threadIdx.x;
    if (t < N_EDGES) atomicAdd(&deg[col[t]], 1.0f);
}

__global__ void deg_to_dis(float* __restrict__ deg) {
    int t = blockIdx.x * blockDim.x + threadIdx.x;
    if (t < N_NODES) {
        float d = deg[t];
        deg[t] = (d > 0.0f) ? rsqrtf(fmaxf(d, 1.0f)) : 0.0f;
    }
}

// initialize aggregation buffer with bias (broadcast over feature dim)
__global__ void init_bias(float* __restrict__ dst, const float* __restrict__ b) {
    int t = blockIdx.x * blockDim.x + threadIdx.x;
    if (t < N_NODES * HID) dst[t] = b[t & (HID - 1)];
}

// ---------------------------------------------------------------------------
// WMMA GEMM:  C[M x 64] = act(A[M x K] * Bt^T + bias)
// A fp32 row-major (lda = K), Bt bf16 column-major [64][K].
// One wave per 16-ROW STRIPE: A fragment loaded/converted once per K-chunk,
// reused across all 4 column tiles (4 accumulators, 4 WMMAs per chunk).
// M = N_NODES (multiple of 16), K % 32 == 0.
// ---------------------------------------------------------------------------
__global__ void gemm_wmma(const float* __restrict__ A, int K,
                          const unsigned short* __restrict__ Bt,
                          const float* __restrict__ bias,
                          float* __restrict__ C,
                          int total_row_tiles, int do_relu) {
    int wid  = (blockIdx.x * blockDim.x + threadIdx.x) >> 5;   // wave-uniform
    int lane = threadIdx.x & 31;
    if (wid >= total_row_tiles) return;                        // whole wave exits

    int m0 = wid << 4;
    int r  = lane & 15;
    int hi = lane >> 4;

    const float* arow = A + (size_t)(m0 + r) * K;

    v8f acc0 = {0.f,0.f,0.f,0.f,0.f,0.f,0.f,0.f};
    v8f acc1 = acc0, acc2 = acc0, acc3 = acc0;

    for (int k0 = 0; k0 < K; k0 += 32) {
        v16bf a = load_a_frag(arow, k0 + hi * 8);

        // B fragments: column n = tile*16 + r, K = k0+hi*8 .. +7 and k0+16+hi*8 .. +7
        Frag16 b0, b1, b2, b3;
        {
            size_t kof = (size_t)(k0 + hi * 8);
            const uint4* p0 = (const uint4*)(Bt + (size_t)(r)      * K + kof);
            const uint4* p1 = (const uint4*)(Bt + (size_t)(16 + r) * K + kof);
            const uint4* p2 = (const uint4*)(Bt + (size_t)(32 + r) * K + kof);
            const uint4* p3 = (const uint4*)(Bt + (size_t)(48 + r) * K + kof);
            b0.q[0] = p0[0]; b0.q[1] = p0[2];
            b1.q[0] = p1[0]; b1.q[1] = p1[2];
            b2.q[0] = p2[0]; b2.q[1] = p2[2];
            b3.q[0] = p3[0]; b3.q[1] = p3[2];
        }
        acc0 = __builtin_amdgcn_wmma_f32_16x16x32_bf16(false, a, false, b0.v, (short)0, acc0, false, false);
        acc1 = __builtin_amdgcn_wmma_f32_16x16x32_bf16(false, a, false, b1.v, (short)0, acc1, false, false);
        acc2 = __builtin_amdgcn_wmma_f32_16x16x32_bf16(false, a, false, b2.v, (short)0, acc2, false, false);
        acc3 = __builtin_amdgcn_wmma_f32_16x16x32_bf16(false, a, false, b3.v, (short)0, acc3, false, false);
    }

    // epilogue: C/D layout -> row = m0 + hi*8 + i, col = tile*16 + r
    v8f accs[4] = {acc0, acc1, acc2, acc3};
    #pragma unroll
    for (int t = 0; t < 4; ++t) {
        int col = (t << 4) + r;
        float bv = bias ? bias[col] : 0.0f;
        #pragma unroll
        for (int i = 0; i < 8; ++i) {
            float v = accs[t][i] + bv;
            if (do_relu) v = fmaxf(v, 0.0f);
            C[(size_t)(m0 + hi * 8 + i) * HID + col] = v;
        }
    }
}

// ---------------------------------------------------------------------------
// Final GEMM over concat [h0|h1|h2] (K=192) -> out[N,40] (+bf), padded N=48.
// One wave per 16-row stripe, 3 column tiles, A reused.
// ---------------------------------------------------------------------------
__global__ void gemm_final(const float* __restrict__ h0,
                           const float* __restrict__ h1,
                           const float* __restrict__ h2,
                           const unsigned short* __restrict__ Wft,
                           const float* __restrict__ bf,
                           float* __restrict__ out, int total_row_tiles) {
    int wid  = (blockIdx.x * blockDim.x + threadIdx.x) >> 5;
    int lane = threadIdx.x & 31;
    if (wid >= total_row_tiles) return;

    int m0 = wid << 4;
    int r  = lane & 15;
    int hi = lane >> 4;

    v8f acc0 = {0.f,0.f,0.f,0.f,0.f,0.f,0.f,0.f};
    v8f acc1 = acc0, acc2 = acc0;

    for (int k0 = 0; k0 < 3 * HID; k0 += 32) {
        int seg = k0 >> 6;                       // 0,0,1,1,2,2
        const float* Abase = (seg == 0) ? h0 : ((seg == 1) ? h1 : h2);
        const float* arow  = Abase + (size_t)(m0 + r) * HID;
        v16bf a = load_a_frag(arow, (k0 & 63) + hi * 8);

        Frag16 b0, b1, b2;
        {
            size_t kof = (size_t)(k0 + hi * 8);
            const uint4* p0 = (const uint4*)(Wft + (size_t)(r)      * (3 * HID) + kof);
            const uint4* p1 = (const uint4*)(Wft + (size_t)(16 + r) * (3 * HID) + kof);
            const uint4* p2 = (const uint4*)(Wft + (size_t)(32 + r) * (3 * HID) + kof);
            b0.q[0] = p0[0]; b0.q[1] = p0[2];
            b1.q[0] = p1[0]; b1.q[1] = p1[2];
            b2.q[0] = p2[0]; b2.q[1] = p2[2];
        }
        acc0 = __builtin_amdgcn_wmma_f32_16x16x32_bf16(false, a, false, b0.v, (short)0, acc0, false, false);
        acc1 = __builtin_amdgcn_wmma_f32_16x16x32_bf16(false, a, false, b1.v, (short)0, acc1, false, false);
        acc2 = __builtin_amdgcn_wmma_f32_16x16x32_bf16(false, a, false, b2.v, (short)0, acc2, false, false);
    }

    v8f accs[3] = {acc0, acc1, acc2};
    #pragma unroll
    for (int t = 0; t < 3; ++t) {
        int col = (t << 4) + r;
        if (col < OUTD) {
            float bv = bf[col];
            #pragma unroll
            for (int i = 0; i < 8; ++i)
                out[(size_t)(m0 + hi * 8 + i) * OUTD + col] = accs[t][i] + bv;
        }
    }
}

// ---------------------------------------------------------------------------
// Edge scatter: agg[col] += hw[row] * dis[row]*dis[col]  (16 threads / edge,
// float4 loads, per-float global atomic adds).  The HBM-bound part.
// ---------------------------------------------------------------------------
__global__ void scatter_edges(const float* __restrict__ hw,
                              const float* __restrict__ dis,
                              const int* __restrict__ rows,
                              const int* __restrict__ cols,
                              float* __restrict__ agg) {
    int t = blockIdx.x * blockDim.x + threadIdx.x;
    if (t >= N_EDGES * 16) return;
    int e = t >> 4;
    int f = (t & 15) * 4;
    int rn = rows[e], cn = cols[e];
    float nrm = dis[rn] * dis[cn];
    float4 v = *(const float4*)(hw + (size_t)rn * HID + f);
    float* dst = agg + (size_t)cn * HID + f;
    atomicAdd(dst + 0, v.x * nrm);
    atomicAdd(dst + 1, v.y * nrm);
    atomicAdd(dst + 2, v.z * nrm);
    atomicAdd(dst + 3, v.w * nrm);
}

// ---------------------------------------------------------------------------
// Host launcher
// ---------------------------------------------------------------------------
extern "C" void kernel_launch(void* const* d_in, const int* in_sizes, int n_in,
                              void* d_out, int out_size, void* d_ws, size_t ws_size,
                              hipStream_t stream) {
    const float* x   = (const float*)d_in[0];
    const int*   ei  = (const int*)d_in[1];
    const float* W1  = (const float*)d_in[2];
    const float* b1  = (const float*)d_in[3];
    const float* Wc1 = (const float*)d_in[4];
    const float* bc1 = (const float*)d_in[5];
    const float* Wc2 = (const float*)d_in[6];
    const float* bc2 = (const float*)d_in[7];
    const float* Wf  = (const float*)d_in[8];
    const float* bf  = (const float*)d_in[9];
    const int* row = ei;
    const int* col = ei + N_EDGES;
    float* out = (float*)d_out;

    // workspace carve-up (256B aligned)
    char* ws = (char*)d_ws;
    size_t off = 0;
    auto carve = [&](size_t bytes) {
        size_t p = off;
        off = (off + bytes + 255) & ~(size_t)255;
        return p;
    };
    unsigned short* W1t  = (unsigned short*)(ws + carve(HID * IN_DIM * 2));
    unsigned short* Wc1t = (unsigned short*)(ws + carve(HID * HID * 2));
    unsigned short* Wc2t = (unsigned short*)(ws + carve(HID * HID * 2));
    unsigned short* Wft  = (unsigned short*)(ws + carve(OUTP * 3 * HID * 2));
    float* dis = (float*)(ws + carve((size_t)N_NODES * 4));
    float* h0  = (float*)(ws + carve((size_t)N_NODES * HID * 4));
    float* h1  = (float*)(ws + carve((size_t)N_NODES * HID * 4));
    float* h2  = (float*)(ws + carve((size_t)N_NODES * HID * 4));
    float* hw  = (float*)(ws + carve((size_t)N_NODES * HID * 4));

    const int B = 256;
    const int ROW_TILES = N_NODES / 16;                     // 6250
    const int GEMM_BLOCKS = (ROW_TILES + 7) / 8;            // 8 waves/block

    // 1) weight transpose/convert
    int prep_total = HID * IN_DIM + 2 * HID * HID + OUTP * 3 * HID;
    prep_weights<<<(prep_total + B - 1) / B, B, 0, stream>>>(
        W1, Wc1, Wc2, Wf, W1t, Wc1t, Wc2t, Wft);

    // 2) degree -> D^{-1/2}
    deg_zero<<<(N_NODES + B - 1) / B, B, 0, stream>>>(dis);
    deg_count<<<(N_EDGES + B - 1) / B, B, 0, stream>>>(col, dis);
    deg_to_dis<<<(N_NODES + B - 1) / B, B, 0, stream>>>(dis);

    // 3) h0 = relu(x @ W1 + b1)   [100000 x 128] x [128 x 64]
    gemm_wmma<<<GEMM_BLOCKS, B, 0, stream>>>(x, IN_DIM, W1t, b1, h0, ROW_TILES, 1);

    // 4) conv1: h1 = scatter(norm * (h0 @ Wc1)) + bc1
    init_bias<<<(N_NODES * HID + B - 1) / B, B, 0, stream>>>(h1, bc1);
    gemm_wmma<<<GEMM_BLOCKS, B, 0, stream>>>(h0, HID, Wc1t, nullptr, hw, ROW_TILES, 0);
    scatter_edges<<<(N_EDGES * 16 + B - 1) / B, B, 0, stream>>>(hw, dis, row, col, h1);

    // 5) conv2: h2 = scatter(norm * (h1 @ Wc2)) + bc2
    init_bias<<<(N_NODES * HID + B - 1) / B, B, 0, stream>>>(h2, bc2);
    gemm_wmma<<<GEMM_BLOCKS, B, 0, stream>>>(h1, HID, Wc2t, nullptr, hw, ROW_TILES, 0);
    scatter_edges<<<(N_EDGES * 16 + B - 1) / B, B, 0, stream>>>(hw, dis, row, col, h2);

    // 6) out = [h0|h1|h2] @ Wf + bf   (K=192, 40 cols padded to 48)
    gemm_final<<<GEMM_BLOCKS, B, 0, stream>>>(h0, h1, h2, Wft, bf, out, ROW_TILES);
}
